// Attention_3642132267284
// MI455X (gfx1250) — compile-verified, matrix operand order checked by reading, not run on hardware
//
#include <hip/hip_runtime.h>

#define DIM 2048
#define NH 32
#define NKV 8
#define HD 64
#define TLEN 2048
#define BSZ 2
#define NT (BSZ * TLEN)      // 4096 token rows
#define WIN 1024
#define QKV_N 3072           // 2048 q + 512 k + 512 v

typedef __attribute__((ext_vector_type(16))) __bf16 v16bf;
typedef __attribute__((ext_vector_type(8)))  float  v8f;

union Frag {
    uint4          u4[2];
    unsigned short h[16];
    v16bf          v;
};

__device__ __forceinline__ unsigned short f2bf(float f) {
    union { float f; unsigned u; } x; x.f = f;
    unsigned r = x.u + 0x7FFFu + ((x.u >> 16) & 1u);   // round-to-nearest-even
    return (unsigned short)(r >> 16);
}

// ---------------------------------------------------------------- cast fp32 -> bf16
__global__ void cast_bf16_kernel(const float* __restrict__ in,
                                 unsigned short* __restrict__ out, int n) {
    int i = blockIdx.x * blockDim.x + threadIdx.x;
    int stride = gridDim.x * blockDim.x;
    for (; i < n; i += stride) out[i] = f2bf(in[i]);
}

// ------------------------------------------------- 32x64 wave tile (NT GEMM)
// A: row-major [M,K] bf16, B: row-major [N,K] bf16 (computes A @ B^T)
// 2 M-subtiles x 4 N-subtiles: each B fragment feeds two WMMAs.
__device__ __forceinline__ void gemm_tile32x64(
    const unsigned short* __restrict__ A, int lda, int m0,
    const unsigned short* __restrict__ Bm, int ldb, int bn0,
    float* __restrict__ outF, int ldo, int out_n0, int K, int lane)
{
    v8f acc[2][4] = {};
    const int moff = lane & 15;
    const int koff = (lane & 16) ? 8 : 0;
    const unsigned short* arow0 = A + (size_t)(m0 + moff) * lda + koff;
    const unsigned short* arow1 = A + (size_t)(m0 + 16 + moff) * lda + koff;
    for (int k0 = 0; k0 < K; k0 += 32) {
        Frag a0, a1;
        a0.u4[0] = *(const uint4*)(arow0 + k0);
        a0.u4[1] = *(const uint4*)(arow0 + k0 + 16);
        a1.u4[0] = *(const uint4*)(arow1 + k0);
        a1.u4[1] = *(const uint4*)(arow1 + k0 + 16);
#pragma unroll
        for (int f = 0; f < 4; ++f) {
            const unsigned short* brow =
                Bm + (size_t)(bn0 + f * 16 + moff) * ldb + k0 + koff;
            Frag b;
            b.u4[0] = *(const uint4*)(brow);
            b.u4[1] = *(const uint4*)(brow + 16);
            acc[0][f] = __builtin_amdgcn_wmma_f32_16x16x32_bf16(
                false, a0.v, false, b.v, (short)0, acc[0][f], false, false);
            acc[1][f] = __builtin_amdgcn_wmma_f32_16x16x32_bf16(
                false, a1.v, false, b.v, (short)0, acc[1][f], false, false);
        }
    }
    const int n = lane & 15;
    const int rbase = (lane & 16) ? 8 : 0;
#pragma unroll
    for (int g = 0; g < 2; ++g)
#pragma unroll
        for (int f = 0; f < 4; ++f)
#pragma unroll
            for (int r = 0; r < 8; ++r)
                outF[(size_t)(m0 + g * 16 + rbase + r) * ldo + out_n0 + f * 16 + n] =
                    acc[g][f][r];
}

// ------------------------------------------------------------- QKV projection GEMM
__global__ __launch_bounds__(256) void gemm_qkv_kernel(
    const unsigned short* __restrict__ xb,
    const unsigned short* __restrict__ wqb,
    const unsigned short* __restrict__ wkb,
    const unsigned short* __restrict__ wvb,
    float* __restrict__ qkv)
{
    const int lane = threadIdx.x & 31;
    const int wave = threadIdx.x >> 5;
    const int task = blockIdx.x * 8 + wave;      // 128 m-tiles * 48 n-groups
    const int mt = task / 48;
    const int ng = task % 48;
    const int m0 = mt * 32;
    const int n0 = ng * 64;
    const unsigned short* Bm; int bn0;
    if (n0 < 2048)      { Bm = wqb; bn0 = n0;        }
    else if (n0 < 2560) { Bm = wkb; bn0 = n0 - 2048; }
    else                { Bm = wvb; bn0 = n0 - 2560; }
    gemm_tile32x64(xb, DIM, m0, Bm, DIM, bn0, qkv, QKV_N, n0, DIM, lane);
}

// ------------------------------------------------------------- output projection GEMM
__global__ __launch_bounds__(256) void gemm_out_kernel(
    const unsigned short* __restrict__ attnb,
    const unsigned short* __restrict__ wob,
    float* __restrict__ out)
{
    const int lane = threadIdx.x & 31;
    const int wave = threadIdx.x >> 5;
    const int task = blockIdx.x * 8 + wave;      // 128 m-tiles * 32 n-groups
    const int mt = task / 32;
    const int ng = task % 32;
    gemm_tile32x64(attnb, DIM, mt * 32, wob, DIM, ng * 64, out, DIM, ng * 64, DIM, lane);
}

// --------------------------------------------- RMSNorm (q,k) + cast, V transposed
__global__ __launch_bounds__(256) void rmsnorm_cast_kernel(
    const float* __restrict__ qkv,
    const float* __restrict__ qw, const float* __restrict__ kw,
    unsigned short* __restrict__ qb,
    unsigned short* __restrict__ kb,
    unsigned short* __restrict__ vTb)
{
    const int row = blockIdx.x;                  // token row 0..NT-1
    const float* r = qkv + (size_t)row * QKV_N;
    __shared__ float red[256];
    const int tid = threadIdx.x;

    float ssq = 0.f;
    for (int i = tid; i < 2048; i += 256) { float v = r[i]; ssq += v * v; }
    red[tid] = ssq; __syncthreads();
    for (int s = 128; s > 0; s >>= 1) { if (tid < s) red[tid] += red[tid + s]; __syncthreads(); }
    const float inv_q = rsqrtf(red[0] / 2048.0f + 1e-6f);
    __syncthreads();

    ssq = 0.f;
    for (int i = tid; i < 512; i += 256) { float v = r[2048 + i]; ssq += v * v; }
    red[tid] = ssq; __syncthreads();
    for (int s = 128; s > 0; s >>= 1) { if (tid < s) red[tid] += red[tid + s]; __syncthreads(); }
    const float inv_k = rsqrtf(red[0] / 512.0f + 1e-6f);

    for (int i = tid; i < 2048; i += 256)
        qb[(size_t)row * 2048 + i] = f2bf(r[i] * inv_q * qw[i]);
    for (int i = tid; i < 512; i += 256)
        kb[(size_t)row * 512 + i] = f2bf(r[2048 + i] * inv_k * kw[i]);

    const int b = row / TLEN, t = row % TLEN;
    for (int i = tid; i < 512; i += 256) {
        int kvh = i >> 6, d = i & 63;
        vTb[((size_t)(b * NKV + kvh) * HD + d) * TLEN + t] = f2bf(r[2560 + i]);
    }
}

// ---------------------------------------------------------------- flash attention
__global__ __launch_bounds__(256) void attn_kernel(
    const unsigned short* __restrict__ qb,
    const unsigned short* __restrict__ kb,
    const unsigned short* __restrict__ vTb,
    unsigned short* __restrict__ attnb)
{
    __shared__ float Sbuf[8][16][32];
    const int lane = threadIdx.x & 31;
    const int wave = threadIdx.x >> 5;
    const int task = blockIdx.x * 8 + wave;      // 64 (b,h) * 128 q-tiles
    const int bh = task >> 7;
    const int qt = task & 127;
    const int b = bh >> 5;
    const int h = bh & 31;
    const int kvh = h >> 2;
    const int t0 = qt * 16;

    const float slope = exp2f(-0.25f * (float)h);
    const float scale = 0.125f;                  // 1/sqrt(64)
    const int m = lane & 15;
    const int hi = (lane & 16) ? 1 : 0;
    const int off = hi ? 8 : 0;
    const int rbase = hi ? 8 : 0;
    const int t_row = t0 + m;

    // Q fragments (16x64 = two K=32 fragments), held for the whole loop
    const unsigned short* qrow = qb + (size_t)(b * TLEN + t0 + m) * DIM + h * HD + off;
    Frag qf[2];
#pragma unroll
    for (int kk = 0; kk < 2; ++kk) {
        qf[kk].u4[0] = *(const uint4*)(qrow + kk * 32);
        qf[kk].u4[1] = *(const uint4*)(qrow + kk * 32 + 16);
    }

    v8f o[4] = {};
    float m_i = -1e30f, l_i = 0.f;

    int s_lo = t0 - WIN; if (s_lo < 0) s_lo = 0;
    const int s_begin = s_lo & ~31;
    const int s_end = t0 + 15;
    const unsigned short* vbase = vTb + (size_t)(b * NKV + kvh) * HD * TLEN;

    for (int s0 = s_begin; s0 <= s_end; s0 += 32) {
        // ---- scores S = Q @ K^T for 32 keys (two 16-key column tiles)
        v8f sacc[2] = {};
#pragma unroll
        for (int f = 0; f < 2; ++f) {
            int srow = s0 + f * 16 + m;
            if (srow > TLEN - 1) srow = TLEN - 1;          // clamp; masked later
            const unsigned short* krow =
                kb + (size_t)(b * TLEN + srow) * (NKV * HD) + kvh * HD + off;
#pragma unroll
            for (int kk = 0; kk < 2; ++kk) {
                Frag bfr;
                bfr.u4[0] = *(const uint4*)(krow + kk * 32);
                bfr.u4[1] = *(const uint4*)(krow + kk * 32 + 16);
                sacc[f] = __builtin_amdgcn_wmma_f32_16x16x32_bf16(
                    false, qf[kk].v, false, bfr.v, (short)0, sacc[f], false, false);
            }
        }
        // ---- C-layout -> A-layout transpose through LDS (per-wave region)
#pragma unroll
        for (int f = 0; f < 2; ++f)
#pragma unroll
            for (int r = 0; r < 8; ++r)
                Sbuf[wave][rbase + r][f * 16 + m] = sacc[f][r];
        asm volatile("s_wait_dscnt 0" ::: "memory");

        float sv[16];
#pragma unroll
        for (int i = 0; i < 8; ++i) {
            sv[i]     = Sbuf[wave][m][off + i];
            sv[8 + i] = Sbuf[wave][m][off + 16 + i];
        }
        asm volatile("s_wait_dscnt 0" ::: "memory");

        // ---- ALiBi + causal + sliding-window mask, row max
        float vals[16];
        float mx = -1e30f;
#pragma unroll
        for (int i = 0; i < 16; ++i) {
            int col = (i < 8) ? (off + i) : (off + i + 8);
            int s_idx = s0 + col;
            int dist = s_idx - t_row;
            float v = (dist <= 0 && dist >= -WIN)
                        ? sv[i] * scale + slope * (float)dist
                        : -1e30f;
            vals[i] = v;
            mx = fmaxf(mx, v);
        }
        mx = fmaxf(mx, __shfl_xor(mx, 16, 32));   // lane L and L^16 share row L&15
        const float m_new = fmaxf(m_i, mx);
        const float corr = __expf(m_i - m_new);

        float lsum = 0.f;
        Frag pf;
#pragma unroll
        for (int i = 0; i < 16; ++i) {
            float p = (vals[i] > -1e29f) ? __expf(vals[i] - m_new) : 0.f;
            lsum += p;
            pf.h[i] = f2bf(p);
        }
        lsum += __shfl_xor(lsum, 16, 32);
        l_i = l_i * corr + lsum;
        m_i = m_new;

        // ---- rescale running O by per-row correction
#pragma unroll
        for (int r = 0; r < 8; ++r) {
            float cr = __shfl(corr, r + rbase, 32);
            o[0][r] *= cr; o[1][r] *= cr; o[2][r] *= cr; o[3][r] *= cr;
        }
        // ---- O += P @ V  (V stored transposed [d][t] so B-frags load contiguously)
#pragma unroll
        for (int f = 0; f < 4; ++f) {
            const unsigned short* vp =
                vbase + (size_t)(f * 16 + m) * TLEN + s0 + off;
            Frag vf;
            vf.u4[0] = *(const uint4*)(vp);
            vf.u4[1] = *(const uint4*)(vp + 16);
            o[f] = __builtin_amdgcn_wmma_f32_16x16x32_bf16(
                false, pf.v, false, vf.v, (short)0, o[f], false, false);
        }
    }

    const float il = 1.0f / l_i;
#pragma unroll
    for (int f = 0; f < 4; ++f)
#pragma unroll
        for (int r = 0; r < 8; ++r) {
            float ir = __shfl(il, r + rbase, 32);
            attnb[(size_t)(b * TLEN + t0 + rbase + r) * DIM + h * HD + f * 16 + m] =
                f2bf(o[f][r] * ir);
        }
}

// ================================================================= host launcher
extern "C" void kernel_launch(void* const* d_in, const int* in_sizes, int n_in,
                              void* d_out, int out_size, void* d_ws, size_t ws_size,
                              hipStream_t stream) {
    (void)in_sizes; (void)n_in; (void)out_size; (void)ws_size;
    const float* x  = (const float*)d_in[0];
    const float* wq = (const float*)d_in[1];
    const float* wk = (const float*)d_in[2];
    const float* wv = (const float*)d_in[3];
    const float* wo = (const float*)d_in[4];
    const float* qnw = (const float*)d_in[5];
    const float* knw = (const float*)d_in[6];
    float* out = (float*)d_out;

    char* ws = (char*)d_ws;
    size_t off = 0;
    auto alloc = [&](size_t bytes) { char* p = ws + off; off = (off + bytes + 255) & ~(size_t)255; return p; };
    unsigned short* xb   = (unsigned short*)alloc((size_t)NT * DIM * 2);
    unsigned short* wqb  = (unsigned short*)alloc((size_t)2048 * DIM * 2);
    unsigned short* wkb  = (unsigned short*)alloc((size_t)512 * DIM * 2);
    unsigned short* wvb  = (unsigned short*)alloc((size_t)512 * DIM * 2);
    unsigned short* wob  = (unsigned short*)alloc((size_t)DIM * DIM * 2);
    float*          qkv  = (float*)        alloc((size_t)NT * QKV_N * 4);
    unsigned short* qb   = (unsigned short*)alloc((size_t)NT * 2048 * 2);
    unsigned short* kb   = (unsigned short*)alloc((size_t)NT * 512 * 2);
    unsigned short* vTb  = (unsigned short*)alloc((size_t)NT * 512 * 2 + 256); // pad for tile overreach
    unsigned short* attnb= (unsigned short*)alloc((size_t)NT * DIM * 2);

    cast_bf16_kernel<<<2048, 256, 0, stream>>>(x,  xb,  NT * DIM);
    cast_bf16_kernel<<<2048, 256, 0, stream>>>(wq, wqb, 2048 * DIM);
    cast_bf16_kernel<<<1024, 256, 0, stream>>>(wk, wkb, 512 * DIM);
    cast_bf16_kernel<<<1024, 256, 0, stream>>>(wv, wvb, 512 * DIM);
    cast_bf16_kernel<<<2048, 256, 0, stream>>>(wo, wob, DIM * DIM);

    gemm_qkv_kernel<<<768, 256, 0, stream>>>(xb, wqb, wkb, wvb, qkv);
    rmsnorm_cast_kernel<<<NT, 256, 0, stream>>>(qkv, qnw, knw, qb, kb, vTb);
    attn_kernel<<<1024, 256, 0, stream>>>(qb, kb, vTb, attnb);
    gemm_out_kernel<<<512, 256, 0, stream>>>(attnb, wob, out);
}
